// SelfAttentionV2_10411000726264
// MI455X (gfx1250) — matrix operand order repaired and checked by reading, not compiled, not run on hardware
//
#include <hip/hip_runtime.h>
#include <stdint.h>

#define SEQ  4096
#define DIN  1024
#define DOUT 1024

typedef __attribute__((ext_vector_type(16))) __bf16 v16bf;
typedef __attribute__((ext_vector_type(8)))  float  v8f;

#if __has_builtin(__builtin_amdgcn_tensor_load_to_lds)
#define USE_TDM 1
typedef unsigned int v4u  __attribute__((ext_vector_type(4)));
typedef int          v8i_ __attribute__((ext_vector_type(8)));
typedef int          v4i_ __attribute__((ext_vector_type(4)));
#endif

union Frag {
  v16bf    v;
  uint16_t h[16];
  uint2    d2[4];
  uint4    d4[2];
};

__device__ __forceinline__ uint16_t f32_to_bf16(float f) {
  uint32_t u = __float_as_uint(f);
  u += 0x7fffu + ((u >> 16) & 1u);   // round-to-nearest-even
  return (uint16_t)(u >> 16);
}

// ---------------------------------------------------------------------------
// fp32 -> bf16 elementwise (bandwidth trivial: ~29 MB total at 23.3 TB/s)
// ---------------------------------------------------------------------------
__global__ void convert_f32_bf16(const float* __restrict__ in,
                                 uint16_t* __restrict__ out, int n) {
  int i = blockIdx.x * blockDim.x + threadIdx.x;
  if (i < n) out[i] = f32_to_bf16(in[i]);
}

// ---------------------------------------------------------------------------
// Projection GEMM: C = X(bf16)[SEQ x DIN] * W(bf16)[DIN x DOUT] -> bf16
// Each wave owns a 16(M) x 64(N) strip: A loaded once per K-step, 4 WMMAs.
// transpose_out==1 stores C transposed ([DOUT x SEQ]) -> produces Kt.
// ---------------------------------------------------------------------------
__global__ __launch_bounds__(256)
void gemm_qkv(const uint16_t* __restrict__ X, const uint16_t* __restrict__ W,
              uint16_t* __restrict__ out, int transpose_out) {
  const int lane  = threadIdx.x & 31;
  const int wave  = threadIdx.x >> 5;
  const int strip = blockIdx.x * 8 + wave;   // 4096 strips = (SEQ/16)*(DOUT/64)
  const int mt    = strip >> 4;              // 0..255
  const int n0    = (strip & 15) * 64;       // col base
  const int half  = lane >> 4;
  const int lm    = lane & 15;
  const int mbase = mt * 16;

  v8f acc[4];
#pragma unroll
  for (int t = 0; t < 4; ++t)
#pragma unroll
    for (int r = 0; r < 8; ++r) acc[t][r] = 0.0f;

  for (int k = 0; k < DIN; k += 32) {
    // A fragment: 16x32 bf16, row m = lane%16, K-halves per ISA layout
    Frag a;
    const uint16_t* ar = X + (size_t)(mbase + lm) * DIN + k + half * 8;
    a.d2[0] = *(const uint2*)(ar + 0);
    a.d2[1] = *(const uint2*)(ar + 4);
    a.d2[2] = *(const uint2*)(ar + 16);
    a.d2[3] = *(const uint2*)(ar + 20);
    // B fragments: lane L holds column n=L%16, K in table order (matches A)
    const uint16_t* br = W + (size_t)(k + half * 8) * DOUT + n0 + lm;
#pragma unroll
    for (int t = 0; t < 4; ++t) {
      Frag b;
      for (int s = 0; s < 16; ++s)
        b.h[s] = br[(size_t)((s & 7) + (s >> 3) * 16) * DOUT + t * 16];
      acc[t] = __builtin_amdgcn_wmma_f32_16x16x32_bf16(
          false, a.v, false, b.v, (short)0, acc[t], false, false);
    }
  }

#pragma unroll
  for (int t = 0; t < 4; ++t)
#pragma unroll
    for (int r = 0; r < 8; ++r) {
      uint16_t hv = f32_to_bf16(acc[t][r]);
      int m = mbase + r + 8 * half;
      int n = n0 + t * 16 + lm;
      if (!transpose_out) out[(size_t)m * DOUT + n] = hv;
      else                out[(size_t)n * SEQ  + m] = hv;
    }
}

// ---------------------------------------------------------------------------
// Flash attention: workgroup = 16 query rows, 8 waves.
//   Q rows staged in LDS via the Tensor Data Mover (one 16x1024 bf16 tile).
//   Per 128-key block:
//     stage1: wave w -> 16x16 score tile vs keys [kb+16w, kb+16w+16)  (32 WMMA)
//     online softmax: row max/sum combined across waves via LDS
//     stage2: P(16x128 bf16) in LDS; wave w accumulates O slice cols
//             [128w,128w+128) via P*V                                  (32 WMMA)
// ---------------------------------------------------------------------------
__global__ __launch_bounds__(256)
void flash_attn(const uint16_t* __restrict__ Q,   // [SEQ][DOUT] bf16
                const uint16_t* __restrict__ Kt,  // [DOUT][SEQ] bf16 (K^T)
                const uint16_t* __restrict__ V,   // [SEQ][DOUT] bf16
                float* __restrict__ O) {          // [SEQ][DOUT] f32
  __shared__ uint16_t Qs[16 * DIN];     // 32 KB
  __shared__ uint16_t Pl[16 * 128];     // 4 KB
  __shared__ float    redM[8][16];
  __shared__ float    redS[8][16];

  const int lane = threadIdx.x & 31;
  const int wave = threadIdx.x >> 5;
  const int half = lane >> 4;
  const int lm   = lane & 15;
  const int qb   = blockIdx.x * 16;

  // scores scaled by 1/sqrt(D) and by log2(e) so we can use exp2
  const float CSCALE = 1.4426950408889634f / 32.0f;

  // ---- stage Q block (16 x 1024 bf16 = 32 KB) into LDS ----
#if defined(USE_TDM)
  if (wave == 0) {
    // Tensor DMA descriptor (CDNA5 ISA): 2-D tile,
    // data_size=2B, tile 1024x16, row stride 1024 elements.
    uint64_t ga  = (uint64_t)(uintptr_t)(Q + (size_t)qb * DOUT);
    uint32_t lds = (uint32_t)(uintptr_t)(&Qs[0]);
    v4u g0;
    g0.x = 1u;                                               // count=1, user mode
    g0.y = lds;                                              // lds_addr
    g0.z = (uint32_t)ga;                                     // global_addr[31:0]
    g0.w = ((uint32_t)(ga >> 32) & 0x01FFFFFFu) | 0x80000000u; // addr[56:32]|type=2
    v8i_ g1;
    g1[0] = 0x00010000;            // workgroup_mask=0, data_size=1 (2 bytes)
    g1[1] = (int)(1024u << 16);    // tensor_dim0 = 1024 (bits 79:48)
    g1[2] = (int)(16u   << 16);    // tensor_dim1 = 16   (bits 111:80)
    g1[3] = (int)(1024u << 16);    // tile_dim0   = 1024 (bits 127:112)
    g1[4] = 16;                    // tile_dim1 = 16, tile_dim2 = 0 (2-D)
    g1[5] = 1024;                  // tensor_dim0_stride low 32
    g1[6] = 0;                     // stride hi / tensor_dim1_stride (unused, 2-D)
    g1[7] = 0;
    v4i_ z4 = {0, 0, 0, 0};
#if __has_include(<hip/amd_detail/amd_gfx1250_TDM.h>)
    v8i_ z8 = {0, 0, 0, 0, 0, 0, 0, 0};
    __builtin_amdgcn_tensor_load_to_lds(g0, g1, z4, z4, z8, 0);
#else
    __builtin_amdgcn_tensor_load_to_lds(g0, g1, z4, z4, 0);
#endif
#if __has_builtin(__builtin_amdgcn_s_wait_tensorcnt)
    __builtin_amdgcn_s_wait_tensorcnt(0);
#endif
  }
#else
  {
    const uint4* src = (const uint4*)(Q + (size_t)qb * DOUT);
    uint4* dst = (uint4*)Qs;
#pragma unroll
    for (int i = 0; i < 8; ++i) dst[threadIdx.x + i * 256] = src[threadIdx.x + i * 256];
  }
#endif
  __syncthreads();

  v8f acc[8];
  float mrow[8], lrow[8];
#pragma unroll
  for (int t = 0; t < 8; ++t)
#pragma unroll
    for (int r = 0; r < 8; ++r) acc[t][r] = 0.0f;
#pragma unroll
  for (int r = 0; r < 8; ++r) { mrow[r] = -1e30f; lrow[r] = 0.0f; }

  for (int kb = 0; kb < SEQ; kb += 128) {
    if (kb + 128 < SEQ) {
      __builtin_prefetch(V + (size_t)(kb + 128) * DOUT + wave * 128, 0, 3);
      __builtin_prefetch(Kt + (size_t)lane * SEQ + kb + 128 + wave * 16, 0, 3);
    }

    // ---- stage 1: S = Q * K^T for this wave's 16-key tile ----
    v8f sc;
#pragma unroll
    for (int r = 0; r < 8; ++r) sc[r] = 0.0f;
    for (int d0 = 0; d0 < DIN; d0 += 32) {
      Frag a;
      const uint16_t* ar = Qs + lm * DIN + d0 + half * 8;
      a.d2[0] = *(const uint2*)(ar + 0);
      a.d2[1] = *(const uint2*)(ar + 4);
      a.d2[2] = *(const uint2*)(ar + 16);
      a.d2[3] = *(const uint2*)(ar + 20);
      Frag b;  // B[k][n] = Kt[d0+k][kb + 16*wave + n]; lane L = column n=L%16, K in table order
      const uint16_t* br = Kt + (size_t)(d0 + half * 8) * SEQ + kb + wave * 16 + lm;
      for (int s = 0; s < 16; ++s)
        b.h[s] = br[(size_t)((s & 7) + (s >> 3) * 16) * SEQ];
      sc = __builtin_amdgcn_wmma_f32_16x16x32_bf16(
          false, a.v, false, b.v, (short)0, sc, false, false);
    }

    // ---- online softmax: per-tile row max, combined across 8 waves ----
    float s2[8], p[8];
#pragma unroll
    for (int r = 0; r < 8; ++r) {
      s2[r] = sc[r] * CSCALE;
      float t = s2[r];
      t = fmaxf(t, __shfl_xor(t, 1));
      t = fmaxf(t, __shfl_xor(t, 2));
      t = fmaxf(t, __shfl_xor(t, 4));
      t = fmaxf(t, __shfl_xor(t, 8));   // max over the 16 lanes of this row group
      if (lm == 0) redM[wave][r + 8 * half] = t;
    }
    __syncthreads();

    float alpha[8];
#pragma unroll
    for (int r = 0; r < 8; ++r) {
      int row = r + 8 * half;
      float bm = redM[0][row];
#pragma unroll
      for (int w = 1; w < 8; ++w) bm = fmaxf(bm, redM[w][row]);
      float mn = fmaxf(mrow[r], bm);
      alpha[r] = exp2f(mrow[r] - mn);
      mrow[r]  = mn;
    }

#pragma unroll
    for (int r = 0; r < 8; ++r) {
      p[r] = exp2f(s2[r] - mrow[r]);
      float t = p[r];
      t += __shfl_xor(t, 1);
      t += __shfl_xor(t, 2);
      t += __shfl_xor(t, 4);
      t += __shfl_xor(t, 8);
      if (lm == 0) redS[wave][r + 8 * half] = t;
      Pl[(r + 8 * half) * 128 + wave * 16 + lm] = f32_to_bf16(p[r]);
    }
    __syncthreads();

#pragma unroll
    for (int r = 0; r < 8; ++r) {
      int row = r + 8 * half;
      float bs = 0.0f;
#pragma unroll
      for (int w = 0; w < 8; ++w) bs += redS[w][row];
      lrow[r] = lrow[r] * alpha[r] + bs;
#pragma unroll
      for (int t = 0; t < 8; ++t) acc[t][r] *= alpha[r];
    }

    // ---- stage 2: O_slice += P(16x128) * V[kb:kb+128, 128w:128w+128) ----
    for (int kk = 0; kk < 128; kk += 32) {
      Frag a;
      const uint16_t* ar = Pl + lm * 128 + kk + half * 8;
      a.d2[0] = *(const uint2*)(ar + 0);
      a.d2[1] = *(const uint2*)(ar + 4);
      a.d2[2] = *(const uint2*)(ar + 16);
      a.d2[3] = *(const uint2*)(ar + 20);
      const uint16_t* br = V + (size_t)(kb + kk + half * 8) * DOUT + wave * 128 + lm;
#pragma unroll
      for (int t = 0; t < 8; ++t) {
        Frag b;
        for (int s = 0; s < 16; ++s)
          b.h[s] = br[(size_t)((s & 7) + (s >> 3) * 16) * DOUT + t * 16];
        acc[t] = __builtin_amdgcn_wmma_f32_16x16x32_bf16(
            false, a.v, false, b.v, (short)0, acc[t], false, false);
      }
    }
    __syncthreads();   // protect Pl/redM/redS before next iteration
  }

  // ---- epilogue: O = acc / l ----
#pragma unroll
  for (int r = 0; r < 8; ++r) {
    float inv = 1.0f / lrow[r];
    int m = qb + r + 8 * half;
#pragma unroll
    for (int t = 0; t < 8; ++t)
      O[(size_t)m * DOUT + wave * 128 + t * 16 + lm] = acc[t][r] * inv;
  }
}

// ---------------------------------------------------------------------------
extern "C" void kernel_launch(void* const* d_in, const int* in_sizes, int n_in,
                              void* d_out, int out_size, void* d_ws, size_t ws_size,
                              hipStream_t stream) {
  const float* X  = (const float*)d_in[0];
  const float* Wq = (const float*)d_in[1];
  const float* Wk = (const float*)d_in[2];
  const float* Wv = (const float*)d_in[3];
  float* O = (float*)d_out;

  char* ws = (char*)d_ws;
  uint16_t* Xb  = (uint16_t*)(ws + 0);          //  8 MB  X  bf16
  uint16_t* Wqb = (uint16_t*)(ws + 8388608);    //  2 MB
  uint16_t* Wkb = (uint16_t*)(ws + 10485760);   //  2 MB
  uint16_t* Wvb = (uint16_t*)(ws + 12582912);   //  2 MB
  uint16_t* Qb  = (uint16_t*)(ws + 14680064);   //  8 MB  Q  bf16
  uint16_t* Ktb = (uint16_t*)(ws + 23068672);   //  8 MB  K^T bf16
  uint16_t* Vb  = (uint16_t*)(ws + 31457280);   //  8 MB  V  bf16

  convert_f32_bf16<<<(SEQ * DIN + 255) / 256, 256, 0, stream>>>(X,  Xb,  SEQ * DIN);
  convert_f32_bf16<<<(DIN * DOUT + 255) / 256, 256, 0, stream>>>(Wq, Wqb, DIN * DOUT);
  convert_f32_bf16<<<(DIN * DOUT + 255) / 256, 256, 0, stream>>>(Wk, Wkb, DIN * DOUT);
  convert_f32_bf16<<<(DIN * DOUT + 255) / 256, 256, 0, stream>>>(Wv, Wvb, DIN * DOUT);

  gemm_qkv<<<512, 256, 0, stream>>>(Xb, Wqb, Qb, 0);   // Q
  gemm_qkv<<<512, 256, 0, stream>>>(Xb, Wkb, Ktb, 1);  // K^T
  gemm_qkv<<<512, 256, 0, stream>>>(Xb, Wvb, Vb, 0);   // V

  flash_attn<<<SEQ / 16, 256, 0, stream>>>(Qb, Ktb, Vb, O);
}
